// B_Attention_ConvNN_K_All_20435454394608
// MI455X (gfx1250) — matrix operand order repaired and checked by reading, not compiled, not run on hardware
//
#include <hip/hip_runtime.h>
#include <hip/hip_bf16.h>

typedef __attribute__((ext_vector_type(16))) _Float16 v16h;
typedef __attribute__((ext_vector_type(8)))  _Float16 v8h;
typedef __attribute__((ext_vector_type(2)))  _Float16 v2h;
typedef __attribute__((ext_vector_type(8)))  float    v8f;

#define HEADS 4
#define KNN 9
#define NTOK 256   // tokens per image after unshuffle (16x16)
#define BATCH 256

// A-fragment K-permutation: storing row K-values at p(kk) makes each lane's
// 16-element A fragment contiguous at row offset hi*16 (ISA 7.12.2 layout:
// frag elem e <-> K = e + 8*(e>=8) + 8*hi).
__device__ __forceinline__ int apos(int kk) {
    return (kk & 7) | ((kk & 8) << 1) | ((kk & 16) >> 1);
}

__device__ __forceinline__ v8f wmma32(v16h a, v16h b, v8f c) {
    return __builtin_amdgcn_wmma_f32_16x16x32_f16(false, a, false, b, (short)0, c,
                                                  false, false);
}

__device__ __forceinline__ v2h pk2(float x, float y) {
    v2h r; r[0] = (_Float16)x; r[1] = (_Float16)y; return r;   // v_cvt_pk_f16_f32
}

// ---------------------------------------------------------------------------
// Kernel 1: per-token local / q / k / v projections.
// LAYER1: input x [B,3,32,32], token load folds pixel_unshuffle(2).
// else:   input pre1 [B*N*64] token-major (shuffle/unshuffle cancel out).
// ---------------------------------------------------------------------------
template<int CU, int R, int D, bool LAYER1>
__global__ __launch_bounds__(256)
void qkvl_kernel(const float* __restrict__ xin,
                 const float* __restrict__ Wc, const float* __restrict__ bc,
                 const float* __restrict__ Wq, const float* __restrict__ Wk,
                 const float* __restrict__ Wv,
                 float* __restrict__ qo, float* __restrict__ ko,
                 float* __restrict__ vo, float* __restrict__ cat)
{
    __shared__ float sWc[CU * R];
    __shared__ float sWq[CU * D];
    __shared__ float sWk[CU * D];
    __shared__ float sWv[CU * D];
    __shared__ float sbc[R];

    const int b = blockIdx.x;
    const int n = threadIdx.x;          // one thread per token, N = 256

    for (int i = threadIdx.x; i < CU * R; i += 256) sWc[i] = Wc[i];
    for (int i = threadIdx.x; i < CU * D; i += 256) {
        sWq[i] = Wq[i]; sWk[i] = Wk[i]; sWv[i] = Wv[i];
    }
    if (threadIdx.x < R) sbc[threadIdx.x] = bc[threadIdx.x];
    __syncthreads();

    float t[CU];
    if (LAYER1) {
        const int y = n >> 4, xp = n & 15;
#pragma unroll
        for (int cu = 0; cu < CU; ++cu) {
            const int c = cu >> 2, sy = (cu >> 1) & 1, sx = cu & 1;
            t[cu] = xin[((size_t)(b * 3 + c) * 32 + (2 * y + sy)) * 32 + (2 * xp + sx)];
        }
    } else {
        const float* p = xin + ((size_t)b * NTOK + n) * CU;
#pragma unroll
        for (int cu = 0; cu < CU; ++cu) t[cu] = p[cu];
    }

    const size_t base = (size_t)b * NTOK + n;
    float* catp = cat + base * (R + D);
    for (int r = 0; r < R; ++r) {
        float acc = sbc[r];
        for (int cu = 0; cu < CU; ++cu) acc += t[cu] * sWc[cu * R + r];
        catp[r] = fmaxf(acc, 0.0f);
    }
    float* qp = qo + base * D;
    float* kp = ko + base * D;
    float* vp = vo + base * D;
    for (int dd = 0; dd < D; ++dd) {
        float aq = 0.f, ak = 0.f, av = 0.f;
        for (int cu = 0; cu < CU; ++cu) {
            const float tv = t[cu];
            aq += tv * sWq[cu * D + dd];
            ak += tv * sWk[cu * D + dd];
            av += tv * sWv[cu * D + dd];
        }
        qp[dd] = aq; kp[dd] = ak; vp[dd] = av;
    }
}

// ---------------------------------------------------------------------------
// Kernel 2: sim = q @ k^T via WMMA (K zero-padded to 32), per-row top-9,
// multi-head softmax over the 9 neighbors, aggregation -> cat[..., R:R+D].
// kf is N-major so B fragments are contiguous v16h; qf rows are in natural K order.
// ---------------------------------------------------------------------------
template<int D, int R, int DH>
__global__ __launch_bounds__(256)
void simattn_kernel(const float* __restrict__ q, const float* __restrict__ k,
                    const float* __restrict__ v, float* __restrict__ cat,
                    float scale)
{
    __shared__ alignas(32) _Float16 kf[NTOK][32];   // [token][K], pad K to 32
    __shared__ alignas(32) _Float16 qf[16][32];     // [row][K], natural K order, pad K to 32
    __shared__ float sim[16][NTOK];

    const int b = blockIdx.x;
    const int strip = blockIdx.y;       // 16 strips of 16 rows
    const int tid = threadIdx.x;

    const float* kb = k + (size_t)b * NTOK * D;
    for (int i = tid; i < NTOK * 32; i += 256) {
        const int n = i >> 5, kk = i & 31;
        kf[n][kk] = (kk < D) ? (_Float16)kb[(size_t)n * D + kk] : (_Float16)0.f;
    }
    const float* qb = q + ((size_t)b * NTOK + strip * 16) * D;
    for (int i = tid; i < 16 * 32; i += 256) {
        const int r = i >> 5, kk = i & 31;
        qf[r][kk] = (kk < D) ? (_Float16)qb[(size_t)r * D + kk] : (_Float16)0.f;
    }
    __syncthreads();

    const int wave = tid >> 5, lane = tid & 31;
    const int hi = lane >> 4, l15 = lane & 15;

    const v16h afrag = *(const v16h*)&qf[l15][hi * 16];
    const v16h b0 = *(const v16h*)&kf[(wave * 2 + 0) * 16 + l15][hi * 16];
    const v16h b1 = *(const v16h*)&kf[(wave * 2 + 1) * 16 + l15][hi * 16];
    v8f c0 = {}, c1 = {};
    c0 = wmma32(afrag, b0, c0);
    c1 = wmma32(afrag, b1, c1);
#pragma unroll
    for (int r = 0; r < 8; ++r) {
        sim[r + (hi ? 8 : 0)][(wave * 2 + 0) * 16 + l15] = c0[r];
        sim[r + (hi ? 8 : 0)][(wave * 2 + 1) * 16 + l15] = c1[r];
    }
    __syncthreads();

    if (tid < 16) {
        const int row = tid;
        const int ntok = strip * 16 + row;

        // top-9 (largest values; stable on ties like lax.top_k)
        float tv[KNN]; int ti[KNN];
#pragma unroll
        for (int i = 0; i < KNN; ++i) { tv[i] = -3.4e38f; ti[i] = 0; }
        for (int m = 0; m < NTOK; ++m) {
            const float s = sim[row][m];
            if (s > tv[KNN - 1]) {
                int p = KNN - 1;
                while (p > 0 && s > tv[p - 1]) { tv[p] = tv[p - 1]; ti[p] = ti[p - 1]; --p; }
                tv[p] = s; ti[p] = m;
            }
        }

        const float* qrow = q + ((size_t)b * NTOK + ntok) * D;
        float qv[D];
        for (int dd = 0; dd < D; ++dd) qv[dd] = qrow[dd];

        float logits[HEADS][KNN];
        for (int kk = 0; kk < KNN; ++kk) {
            const float* krow = k + ((size_t)b * NTOK + ti[kk]) * D;
            for (int h = 0; h < HEADS; ++h) {
                float acc = 0.f;
                for (int dd = 0; dd < DH; ++dd) acc += qv[h * DH + dd] * krow[h * DH + dd];
                logits[h][kk] = acc * scale;
            }
        }
        float agg[D];
        for (int dd = 0; dd < D; ++dd) agg[dd] = 0.f;
        for (int h = 0; h < HEADS; ++h) {
            float mx = logits[h][0];
            for (int kk = 1; kk < KNN; ++kk) mx = fmaxf(mx, logits[h][kk]);
            float w[KNN]; float den = 0.f;
            for (int kk = 0; kk < KNN; ++kk) { w[kk] = __expf(logits[h][kk] - mx); den += w[kk]; }
            const float inv = 1.f / den;
            for (int kk = 0; kk < KNN; ++kk) {
                const float* vrow = v + ((size_t)b * NTOK + ti[kk]) * D;
                const float ww = w[kk] * inv;
                for (int dd = 0; dd < DH; ++dd) agg[h * DH + dd] += ww * vrow[h * DH + dd];
            }
        }
        float* cp = cat + ((size_t)b * NTOK + ntok) * (R + D) + R;
        for (int dd = 0; dd < D; ++dd) cp[dd] = agg[dd];
    }
}

// ---------------------------------------------------------------------------
// Kernel 3: out = cat @ Wo + bo via WMMA, fused store-side index map.
// ca rows in natural K order; Wo transposed to N-major in LDS so B
// fragments are contiguous v16h loads.
// ---------------------------------------------------------------------------
template<int KD, int CO, bool LAYER1>
__global__ __launch_bounds__(256)
void outproj_kernel(const float* __restrict__ cat, const float* __restrict__ Wo,
                    const float* __restrict__ bo, float* __restrict__ out)
{
    constexpr int ROWS = 64;
    constexpr int CT = CO / 16;            // column tiles (4 or 8)
    constexpr int TPW = (4 * CT) / 8;      // tiles per wave (2 or 4)
    constexpr int KS = KD / 32;            // k-steps (1 or 2)

    __shared__ alignas(32) _Float16 ca[ROWS][KD];   // [row][K], natural K order
    __shared__ alignas(32) _Float16 woT[CO][KD];    // N-major
    __shared__ float sb[CO];

    const int b = blockIdx.x;
    const int rb = blockIdx.y;             // 4 row-blocks of 64 tokens
    const int tid = threadIdx.x;

    const float* catp = cat + ((size_t)b * NTOK + rb * ROWS) * KD;
    for (int i = tid; i < ROWS * KD; i += 256) {
        const int row = i / KD, kk = i % KD;
        ca[row][kk] = (_Float16)catp[i];
    }
    for (int i = tid; i < KD * CO; i += 256) {
        const int r = i / CO, c = i % CO;
        woT[c][r] = (_Float16)Wo[i];
    }
    for (int i = tid; i < CO; i += 256) sb[i] = bo[i];
    __syncthreads();

    const int wave = tid >> 5, lane = tid & 31;
    const int hi = lane >> 4, l15 = lane & 15;

    for (int t = 0; t < TPW; ++t) {
        const int tile = wave * TPW + t;
        const int rt = tile / CT, ct = tile % CT;
        v8f acc = {};
#pragma unroll
        for (int ks = 0; ks < KS; ++ks) {
            const v16h afrag = *(const v16h*)&ca[rt * 16 + l15][ks * 32 + hi * 16];
            const v16h bfrag = *(const v16h*)&woT[ct * 16 + l15][ks * 32 + hi * 16];
            acc = wmma32(afrag, bfrag, acc);
        }
#pragma unroll
        for (int r = 0; r < 8; ++r) {
            const int row = rt * 16 + r + (hi ? 8 : 0);
            const int n = rb * ROWS + row;
            const int co = ct * 16 + l15;
            const float val = acc[r] + sb[co];
            if (LAYER1) {
                out[((size_t)b * NTOK + n) * CO + co] = val;
            } else {
                const int y = n >> 4, xx = n & 15;
                const int cp = co >> 2, sy = (co >> 1) & 1, sx = co & 1;
                out[(size_t)b * 32768 + (size_t)cp * 1024 +
                    (size_t)(y * 2 + sy) * 32 + (xx * 2 + sx)] = val;
            }
        }
    }
}

// ---------------------------------------------------------------------------
// Kernel 4: fc1 partial GEMM. a2[256,32768] @ W[32768,1024], K split in 4
// chunks (grid.z) -> partial sums (bias+relu fused in the reduce kernel).
// 128x64 block tile, ping-pong LDS double buffering, register prefetch of the
// next global tile, GL2 prefetch of the strip after next, fragment-layout LDS
// so each WMMA operand is a pair of ds_load_b128s.
// ---------------------------------------------------------------------------
#define FC1_KCHUNK 8192
#define FC1_KSPLIT 4

__global__ __launch_bounds__(256)
void fc1_kernel(const float* __restrict__ a2, const float* __restrict__ w,
                float* __restrict__ h1p)
{
    __shared__ alignas(32) _Float16 at[2][128][32];  // [row][K], natural K order
    __shared__ alignas(32) _Float16 wt[2][64][32];   // N-major
    const int nB = blockIdx.x;   // 16 column blocks
    const int mB = blockIdx.y;   // 2 row blocks
    const int kB = blockIdx.z;   // 4 K chunks
    const int tid = threadIdx.x;
    const int wave = tid >> 5, lane = tid & 31;
    const int hi = lane >> 4, l15 = lane & 15;

    const float* abase = a2 + (size_t)mB * 128 * 32768 + (size_t)kB * FC1_KCHUNK;
    const float* wbase = w + (size_t)kB * FC1_KCHUNK * 1024 + (size_t)nB * 64;

    float2 ra[8];
    float  rw[8];
    auto loadtiles = [&](int k0) {
#pragma unroll
        for (int t = 0; t < 8; ++t) {              // A: 128x32, pair loads
            const int pi = t * 256 + tid;
            const int row = pi >> 4, kk = (pi & 15) * 2;
            ra[t] = *(const float2*)&abase[(size_t)row * 32768 + k0 + kk];
        }
#pragma unroll
        for (int t = 0; t < 4; ++t) {              // W: 32x64, K-pair gathers
            const int pi = t * 256 + tid;
            const int c = pi & 63, kk = (pi >> 6) * 2;
            rw[2 * t]     = wbase[(size_t)(k0 + kk) * 1024 + c];
            rw[2 * t + 1] = wbase[(size_t)(k0 + kk + 1) * 1024 + c];
        }
    };

    v8f acc0 = {}, acc1 = {}, acc2 = {}, acc3 = {};
    loadtiles(0);
    int buf = 0;
    for (int k0 = 0; k0 < FC1_KCHUNK; k0 += 32, buf ^= 1) {
#pragma unroll
        for (int t = 0; t < 8; ++t) {
            const int pi = t * 256 + tid;
            const int row = pi >> 4, kk = (pi & 15) * 2;
            *(v2h*)&at[buf][row][kk] = pk2(ra[t].x, ra[t].y);
        }
#pragma unroll
        for (int t = 0; t < 4; ++t) {
            const int pi = t * 256 + tid;
            const int c = pi & 63, kk = (pi >> 6) * 2;
            *(v2h*)&wt[buf][c][kk] = pk2(rw[2 * t], rw[2 * t + 1]);
        }
        __syncthreads();
        if (k0 + 32 < FC1_KCHUNK) loadtiles(k0 + 32);   // overlap with WMMA
        if (k0 + 64 < FC1_KCHUNK) {                     // warm GL2 for k0+64
            __builtin_prefetch(&abase[(size_t)(tid >> 1) * 32768 + k0 + 64 + (tid & 1) * 16], 0, 1);
            __builtin_prefetch(&wbase[(size_t)(k0 + 64 + (tid >> 3)) * 1024 + (tid & 7) * 8], 0, 1);
        }
        // Hoist all fragment loads ahead of the WMMA chain so the ds_load_b128s
        // pipeline behind a single dscnt wait instead of one wait per WMMA.
        const v16h a  = *(const v16h*)&at[buf][wave * 16 + l15][hi * 16];
        const v16h b0 = *(const v16h*)&wt[buf][0 * 16 + l15][hi * 16];
        const v16h b1 = *(const v16h*)&wt[buf][1 * 16 + l15][hi * 16];
        const v16h b2 = *(const v16h*)&wt[buf][2 * 16 + l15][hi * 16];
        const v16h b3 = *(const v16h*)&wt[buf][3 * 16 + l15][hi * 16];
        acc0 = wmma32(a, b0, acc0);
        acc1 = wmma32(a, b1, acc1);
        acc2 = wmma32(a, b2, acc2);
        acc3 = wmma32(a, b3, acc3);
        // single barrier/iter is safe: next store targets the other buffer,
        // whose last readers finished before they reached this barrier.
    }

    float* hp = h1p + (size_t)kB * 256 * 1024;
    v8f accs[4] = {acc0, acc1, acc2, acc3};
#pragma unroll
    for (int ct = 0; ct < 4; ++ct) {
#pragma unroll
        for (int r = 0; r < 8; ++r) {
            const int m = mB * 128 + wave * 16 + r + (hi ? 8 : 0);
            const int n = nB * 64 + ct * 16 + l15;
            hp[(size_t)m * 1024 + n] = accs[ct][r];
        }
    }
}

// Deterministic partial reduction + bias + relu (no float atomics).
__global__ void fc1_reduce_kernel(const float* __restrict__ h1p,
                                  const float* __restrict__ bias,
                                  float* __restrict__ h1)
{
    const int idx = blockIdx.x * 256 + threadIdx.x;   // 256*1024 elements
    const int n = idx & 1023;
    float s = bias[n];
#pragma unroll
    for (int p = 0; p < FC1_KSPLIT; ++p) s += h1p[(size_t)p * 256 * 1024 + idx];
    h1[idx] = fmaxf(s, 0.f);
}

// ---------------------------------------------------------------------------
// Kernel 5: fc2 — [256,1024] @ [1024,10] + b. Tiny; one thread per output.
// ---------------------------------------------------------------------------
__global__ void fc2_kernel(const float* __restrict__ h1, const float* __restrict__ w,
                           const float* __restrict__ bias, float* __restrict__ out)
{
    const int idx = blockIdx.x * blockDim.x + threadIdx.x;
    if (idx >= BATCH * 10) return;
    const int b = idx / 10, c = idx % 10;
    const float* hp = h1 + (size_t)b * 1024;
    float acc = bias[c];
    for (int i = 0; i < 1024; ++i) acc += hp[i] * w[(size_t)i * 10 + c];
    out[idx] = acc;
}

// ---------------------------------------------------------------------------
extern "C" void kernel_launch(void* const* d_in, const int* in_sizes, int n_in,
                              void* d_out, int out_size, void* d_ws, size_t ws_size,
                              hipStream_t stream)
{
    (void)in_sizes; (void)n_in; (void)out_size; (void)ws_size;
    const float* x    = (const float*)d_in[0];
    const float* w1c  = (const float*)d_in[1];
    const float* b1c  = (const float*)d_in[2];
    const float* w1q  = (const float*)d_in[3];
    const float* w1k  = (const float*)d_in[4];
    const float* w1v  = (const float*)d_in[5];
    const float* w1o  = (const float*)d_in[6];
    const float* b1o  = (const float*)d_in[7];
    const float* w2c  = (const float*)d_in[8];
    const float* b2c  = (const float*)d_in[9];
    const float* w2q  = (const float*)d_in[10];
    const float* w2k  = (const float*)d_in[11];
    const float* w2v  = (const float*)d_in[12];
    const float* w2o  = (const float*)d_in[13];
    const float* b2o  = (const float*)d_in[14];
    const float* fc1w = (const float*)d_in[15];
    const float* fc1b = (const float*)d_in[16];
    const float* fc2w = (const float*)d_in[17];
    const float* fc2b = (const float*)d_in[18];
    float* out = (float*)d_out;
    float* ws  = (float*)d_ws;

    // Workspace layout (floats); a2 overlays the layer-1 scratch (dead by then).
    const size_t QK1 = (size_t)BATCH * NTOK * 16;   // 1,048,576
    const size_t QK2 = (size_t)BATCH * NTOK * 32;   // 2,097,152
    float* q1   = ws + 0;
    float* k1   = q1 + QK1;
    float* v1   = k1 + QK1;
    float* cat1 = v1 + QK1;                          // [B*N*32]
    float* pre1 = cat1 + QK2;                        // [B*N*64] token-major
    float* a2   = ws + 0;                            // overlays q1..pre1
    float* q2   = pre1 + (size_t)BATCH * NTOK * 64;
    float* k2   = q2 + QK2;
    float* v2   = k2 + QK2;
    float* cat2 = v2 + QK2;                          // [B*N*64]
    float* h1   = cat2 + (size_t)BATCH * NTOK * 64;  // [256*1024]
    float* h1p  = h1 + (size_t)256 * 1024;           // [4][256*1024]

    // ---- layer 1 ----
    qkvl_kernel<12, 16, 16, true><<<BATCH, 256, 0, stream>>>(
        x, w1c, b1c, w1q, w1k, w1v, q1, k1, v1, cat1);
    simattn_kernel<16, 16, 4><<<dim3(BATCH, 16), 256, 0, stream>>>(
        q1, k1, v1, cat1, 0.5f);                         // 1/sqrt(4)
    outproj_kernel<32, 64, true><<<dim3(BATCH, 4), 256, 0, stream>>>(
        cat1, w1o, b1o, pre1);

    // ---- layer 2 (pixel_shuffle+unshuffle cancel: pre1 is already tokens) ----
    qkvl_kernel<64, 32, 32, false><<<BATCH, 256, 0, stream>>>(
        pre1, w2c, b2c, w2q, w2k, w2v, q2, k2, v2, cat2);
    simattn_kernel<32, 32, 8><<<dim3(BATCH, 16), 256, 0, stream>>>(
        q2, k2, v2, cat2, 0.35355339059327373f);         // 1/sqrt(8)
    outproj_kernel<64, 128, false><<<dim3(BATCH, 4), 256, 0, stream>>>(
        cat2, w2o, b2o, a2);                             // fused pixel_shuffle+flatten

    // ---- MLP head ----
    fc1_kernel<<<dim3(16, 2, FC1_KSPLIT), 256, 0, stream>>>(a2, fc1w, h1p);
    fc1_reduce_kernel<<<1024, 256, 0, stream>>>(h1p, fc1b, h1);
    fc2_kernel<<<10, 256, 0, stream>>>(h1, fc2w, fc2b, out);
}
